// NodeEmbeddingGraphormer_29008209117682
// MI455X (gfx1250) — matrix operand order repaired and checked by reading, not compiled, not run on hardware
//
#include <hip/hip_runtime.h>
#include <hip/hip_bf16.h>

// Graphormer forward for gfx1250 (MI455X). Bias-bandwidth-bound (134MB f16
// attn bias read/layer @ 23.3TB/s); compute via v_wmma_f32_16x16x32_f16.
// Flash attention stages K/V chunks into LDS with global_load_async_to_lds
// (ASYNCcnt double-buffering) so K/V global traffic is shared by 4 waves.

#define B_   32
#define N_   512
#define D_   256
#define H_   8
#define DH_  32
#define E_   16384
#define L_   3
#define BN_  (B_ * N_)
#define NUM_EMB_ 12

typedef __attribute__((ext_vector_type(16))) _Float16 v16h;
typedef __attribute__((ext_vector_type(8)))  float    v8f;
typedef __attribute__((address_space(3))) _Float16 lds_f16_t;

// ---- WMMA fragment loaders (contiguous k order, used alike by A and B) ----
// A (16x32 f16): lane&15 = row M; hi=lane>>4; elem e -> K=16*hi+e.
static __device__ __forceinline__ v16h a_frag_f32(const float* __restrict__ A,
                                                  int lda, int k0, int lane) {
  const float* p = A + (lane & 15) * lda + k0 + 16 * (lane >> 4);
  v16h a;
#pragma unroll
  for (int e = 0; e < 8; ++e) a[e] = (_Float16)p[e];
#pragma unroll
  for (int e = 0; e < 8; ++e) a[8 + e] = (_Float16)p[8 + e];
  return a;
}
static __device__ __forceinline__ v16h a_frag_f16(const _Float16* __restrict__ A,
                                                  int lda, int k0, int lane) {
  const _Float16* p = A + (lane & 15) * lda + k0 + 16 * (lane >> 4);
  v16h a;
#pragma unroll
  for (int e = 0; e < 8; ++e) a[e] = p[e];
#pragma unroll
  for (int e = 0; e < 8; ++e) a[8 + e] = p[8 + e];
  return a;
}
// B (32x16 f16): lane&15 = col N; elem e -> K = 16*(lane>>4)+e. W is KxN f32.
static __device__ __forceinline__ v16h b_frag_f32(const float* __restrict__ W,
                                                  int ldn, int k0, int col, int lane) {
  const float* p = W + (size_t)(k0 + 16 * (lane >> 4)) * ldn + col + (lane & 15);
  v16h b;
#pragma unroll
  for (int e = 0; e < 16; ++e) b[e] = (_Float16)p[(size_t)e * ldn];
  return b;
}

static __device__ __forceinline__ v8f wmma_f16(v16h a, v16h b, v8f c) {
  return __builtin_amdgcn_wmma_f32_16x16x32_f16(false, a, false, b, (short)0, c,
                                                false, false);
}

// Async DMA of one 32x32 f16 tile (row-major, 64B rows) global -> LDS.
// Wave `wv` copies rows wv*8 .. wv*8+7; one b128 per lane => 1 instruction.
static __device__ __forceinline__ void async_tile_rows(
    const _Float16* __restrict__ gsrc, _Float16* lds_dst, int lane, int wv) {
  int row = wv * 8 + (lane >> 2);
  int colh = (lane & 3) * 8;  // 8 halves = 16B per lane
  const _Float16* g = gsrc + (size_t)row * DH_ + colh;
  unsigned loff = (unsigned)(size_t)(lds_f16_t*)(lds_dst + row * DH_ + colh);
  asm volatile("global_load_async_to_lds_b128 %0, %1, off"
               :: "v"(loff), "v"(g) : "memory");
}

// ------------------------- small VALU kernels -----------------------------
__global__ void __launch_bounds__(256)
embed_kernel(const float* __restrict__ x, const float* __restrict__ xc,
             const float* __restrict__ w, const float* __restrict__ bias,
             float* __restrict__ h) {
  int idx = blockIdx.x * 256 + threadIdx.x;          // BN*D
  int d = idx & (D_ - 1);
  int row = idx >> 8;
  const float* xp = x + row * 3;
  const float* xcp = xc + row * 3;
  float acc = bias[d];
#pragma unroll
  for (int f = 0; f < 3; ++f)
    acc += xp[f] * w[f * D_ + d] + xcp[f] * w[(3 + f) * D_ + d];
  h[idx] = acc;
}

// ev = relu(edge_attr @ ee_w1 + b1) @ ee_w2 + b2   (F_EDGE=1) ; one wave/edge
__global__ void __launch_bounds__(256)
edge_vals(const float* __restrict__ ea, const float* __restrict__ w1,
          const float* __restrict__ b1, const float* __restrict__ w2,
          const float* __restrict__ b2, float* __restrict__ ev) {
  int lane = threadIdx.x & 31;
  int e = blockIdx.x * 8 + (threadIdx.x >> 5);        // B*E edges
  float a = ea[e];
  float acc[H_];
#pragma unroll
  for (int hh = 0; hh < H_; ++hh) acc[hh] = 0.f;
#pragma unroll
  for (int t = 0; t < 8; ++t) {
    int d = lane * 8 + t;
    float r = a * w1[d] + b1[d];
    r = r > 0.f ? r : 0.f;
#pragma unroll
    for (int hh = 0; hh < H_; ++hh) acc[hh] += r * w2[d * H_ + hh];
  }
#pragma unroll
  for (int hh = 0; hh < H_; ++hh) {
#pragma unroll
    for (int off = 1; off < 32; off <<= 1)
      acc[hh] += __shfl_xor(acc[hh], off, 32);
  }
  if (lane == 0) {
#pragma unroll
    for (int hh = 0; hh < H_; ++hh)
      ev[(size_t)e * H_ + hh] = acc[hh] + b2[hh];
  }
}

// attn_bias[b,h,i,j] = dist_emb[clip(spd)][h]  (f16, (B,H,N,N))
__global__ void __launch_bounds__(256)
bias_base(const int* __restrict__ spd, const float* __restrict__ demb,
          _Float16* __restrict__ bias) {
  size_t idx = (size_t)blockIdx.x * 256 + threadIdx.x;   // B*N*N
  int j = (int)(idx & (N_ - 1));
  size_t rem = idx >> 9;
  int i = (int)(rem & (N_ - 1));
  int b = (int)(rem >> 9);
  int s = spd[idx];
  s = s < 0 ? 0 : (s > NUM_EMB_ - 1 ? NUM_EMB_ - 1 : s);
#pragma unroll
  for (int hh = 0; hh < H_; ++hh)
    bias[(((size_t)(b * H_ + hh)) * N_ + i) * N_ + j] = (_Float16)demb[s * H_ + hh];
}

// scatter-set: bias[b,h,src,dst] = spd_bias + ev  (matches .at[].set + add)
__global__ void __launch_bounds__(256)
edge_scatter(const int* __restrict__ eidx, const int* __restrict__ spd,
             const float* __restrict__ demb, const float* __restrict__ ev,
             _Float16* __restrict__ bias) {
  int t = blockIdx.x * 256 + threadIdx.x;               // B*E
  int b = t / E_, e = t - b * E_;
  int src = eidx[(size_t)b * 2 * E_ + e];
  int dst = eidx[(size_t)b * 2 * E_ + E_ + e];
  int s = spd[((size_t)b * N_ + src) * N_ + dst];
  s = s < 0 ? 0 : (s > NUM_EMB_ - 1 ? NUM_EMB_ - 1 : s);
#pragma unroll
  for (int hh = 0; hh < H_; ++hh)
    bias[(((size_t)(b * H_ + hh)) * N_ + src) * N_ + dst] =
        (_Float16)(demb[s * H_ + hh] + ev[(size_t)t * H_ + hh]);
}

__global__ void __launch_bounds__(256)
mask_copy(const float* __restrict__ h, const int* __restrict__ mask,
          float* __restrict__ out) {
  int idx = blockIdx.x * 256 + threadIdx.x;             // BN*D
  int row = idx >> 8;
  out[idx] = mask[row] ? h[idx] : 0.f;
}

// residual + LayerNorm, one wave per row (D=256 -> 8 elems/lane)
__global__ void __launch_bounds__(256)
ln_res(const float* __restrict__ res, float* __restrict__ h,
       const float* __restrict__ g, const float* __restrict__ bb) {
  int lane = threadIdx.x & 31;
  int row = blockIdx.x * 8 + (threadIdx.x >> 5);
  const float* hr = h + (size_t)row * D_;
  const float* rr = res + (size_t)row * D_;
  float xv[8];
  float s = 0.f;
#pragma unroll
  for (int i = 0; i < 8; ++i) {
    xv[i] = hr[lane + 32 * i] + rr[lane + 32 * i];
    s += xv[i];
  }
#pragma unroll
  for (int off = 1; off < 32; off <<= 1) s += __shfl_xor(s, off, 32);
  float mean = s * (1.f / (float)D_);
  float v = 0.f;
#pragma unroll
  for (int i = 0; i < 8; ++i) { float d = xv[i] - mean; v += d * d; }
#pragma unroll
  for (int off = 1; off < 32; off <<= 1) v += __shfl_xor(v, off, 32);
  float inv = rsqrtf(v * (1.f / (float)D_) + 1e-5f);
#pragma unroll
  for (int i = 0; i < 8; ++i) {
    int c = lane + 32 * i;
    h[(size_t)row * D_ + c] = (xv[i] - mean) * inv * g[c] + bb[c];
  }
}

// ---------------- WMMA GEMM: C[M,N] = A[M,K] @ W[K,N] + bias --------------
// Wave tile 16x64 (A-fragment reused across 4 B-fragments / 4 WMMAs per K32).
enum { EPI_STORE = 0, EPI_RELU = 1, EPI_QKV = 2 };

template <int EPI>
__global__ void __launch_bounds__(256)
gemm16(const float* __restrict__ A, const float* __restrict__ W,
       const float* __restrict__ bias, void* __restrict__ out, int K, int Ncols) {
  int lane = threadIdx.x & 31;
  int wid = blockIdx.x * 8 + (threadIdx.x >> 5);
  int ntn = Ncols >> 6;
  int tm = wid / ntn, tn = wid - tm * ntn;
  if (tm >= (BN_ >> 4)) return;
  int row0 = tm << 4, col0 = tn << 6;
  const float* Arow = A + (size_t)row0 * K;
  v8f acc[4];
#pragma unroll
  for (int t = 0; t < 4; ++t) acc[t] = (v8f){};
  for (int k0 = 0; k0 < K; k0 += 32) {
    if (k0 + 32 < K)
      __builtin_prefetch(Arow + (lane & 15) * K + k0 + 32, 0, 1);
    v16h a = a_frag_f32(Arow, K, k0, lane);
#pragma unroll
    for (int t = 0; t < 4; ++t) {
      v16h b = b_frag_f32(W, Ncols, k0, col0 + 16 * t, lane);
      acc[t] = wmma_f16(a, b, acc[t]);
    }
  }
  int rbase = 8 * (lane >> 4);
  int n0 = lane & 15;
#pragma unroll
  for (int t = 0; t < 4; ++t) {
    int col = col0 + 16 * t + n0;
    float bv = bias[col];
#pragma unroll
    for (int j = 0; j < 8; ++j) {
      int row = row0 + rbase + j;
      float v = acc[t][j] + bv;
      if (EPI == EPI_RELU) v = v > 0.f ? v : 0.f;
      if (EPI == EPI_QKV) {
        int b = row >> 9, node = row & (N_ - 1);
        int head = col >> 5, d = col & (DH_ - 1);
        ((_Float16*)out)[(((size_t)(b * H_ + head)) * N_ + node) * DH_ + d] =
            (_Float16)v;
      } else {
        ((float*)out)[(size_t)row * Ncols + col] = v;
      }
    }
  }
}

// ---- flash attention: 4-wave block owns (b, head, 4 q-tiles); K/V chunks
// ---- double-buffered in LDS via global_load_async_to_lds (ASYNCcnt).
__global__ void __launch_bounds__(128)
flash_attn(const _Float16* __restrict__ Q, const _Float16* __restrict__ Kt,
           const _Float16* __restrict__ V, const _Float16* __restrict__ Bt,
           const int* __restrict__ mask, float* __restrict__ outp, float scale) {
  __shared__ _Float16 kbuf[2][32 * DH_];   // 32 keys x 32 dh per chunk
  __shared__ _Float16 vbuf[2][32 * DH_];
  __shared__ _Float16 pbuf[4][16 * 40];    // per-wave P-transpose tile

  int lane = threadIdx.x & 31;
  int w = threadIdx.x >> 5;
  int blk = blockIdx.x;                    // B*H*8 blocks
  int b = blk / (H_ * 8);
  int rem = blk - b * H_ * 8;
  int head = rem >> 3;
  int qtile = (rem & 7) * 4 + w;
  int row0 = qtile << 4;
  size_t bh = (size_t)(b * H_ + head);
  const _Float16* Qb = Q + bh * N_ * DH_;
  const _Float16* Kb = Kt + bh * N_ * DH_;
  const _Float16* Vb = V + bh * N_ * DH_;
  const _Float16* Bb = Bt + bh * N_ * N_;
  int n0 = lane & 15, hi = lane >> 4;

  v16h qa = a_frag_f16(Qb + (size_t)row0 * DH_, DH_, 0, lane);
  v8f o0 = (v8f){}, o1 = (v8f){};
  float m[8], l[8];
#pragma unroll
  for (int j = 0; j < 8; ++j) { m[j] = -1e30f; l[j] = 0.f; }

  // prologue: DMA chunk 0 into buffer 0
  async_tile_rows(Kb, kbuf[0], lane, w);
  async_tile_rows(Vb, vbuf[0], lane, w);

  const int nch = N_ / 32;                 // 16 chunks
  for (int c = 0; c < nch; ++c) {
    int buf = c & 1;
    if (c + 1 < nch) {                     // prefetch next chunk, other buffer
      async_tile_rows(Kb + (size_t)(c + 1) * 32 * DH_, kbuf[buf ^ 1], lane, w);
      async_tile_rows(Vb + (size_t)(c + 1) * 32 * DH_, vbuf[buf ^ 1], lane, w);
      asm volatile("s_wait_asynccnt 0x2" ::: "memory");  // current chunk done
    } else {
      asm volatile("s_wait_asynccnt 0x0" ::: "memory");
    }
    __syncthreads();
    int j0 = c * 32;
    const _Float16* Kl = kbuf[buf];
    const _Float16* Vl = vbuf[buf];

    // K as B-operand from LDS: B[k][n] = K[j0+n][k]
    v16h kb0, kb1;
    {
      const _Float16* p0 = Kl + n0 * DH_ + 16 * hi;
      const _Float16* p1 = Kl + (16 + n0) * DH_ + 16 * hi;
#pragma unroll
      for (int e = 0; e < 16; ++e) { kb0[e] = p0[e]; kb1[e] = p1[e]; }
    }
    v8f z = (v8f){};
    v8f s0 = wmma_f16(qa, kb0, z);
    v8f s1 = wmma_f16(qa, kb1, z);
    float mk0 = mask[b * N_ + j0 + n0] ? 0.f : -1e30f;
    float mk1 = mask[b * N_ + j0 + 16 + n0] ? 0.f : -1e30f;
    float p0v[8], p1v[8];
#pragma unroll
    for (int j = 0; j < 8; ++j) {
      int row = row0 + 8 * hi + j;
      s0[j] = s0[j] * scale + (float)Bb[(size_t)row * N_ + j0 + n0] + mk0;
      s1[j] = s1[j] * scale + (float)Bb[(size_t)row * N_ + j0 + 16 + n0] + mk1;
    }
#pragma unroll
    for (int j = 0; j < 8; ++j) {
      float cm = fmaxf(s0[j], s1[j]);
      cm = fmaxf(cm, __shfl_xor(cm, 1, 32));
      cm = fmaxf(cm, __shfl_xor(cm, 2, 32));
      cm = fmaxf(cm, __shfl_xor(cm, 4, 32));
      cm = fmaxf(cm, __shfl_xor(cm, 8, 32));
      float mn = fmaxf(m[j], cm);
      float corr = __expf(m[j] - mn);
      m[j] = mn;
      float e0 = __expf(s0[j] - mn), e1 = __expf(s1[j] - mn);
      float rs = e0 + e1;
      rs += __shfl_xor(rs, 1, 32);
      rs += __shfl_xor(rs, 2, 32);
      rs += __shfl_xor(rs, 4, 32);
      rs += __shfl_xor(rs, 8, 32);
      l[j] = l[j] * corr + rs;
      o0[j] *= corr; o1[j] *= corr;
      p0v[j] = e0; p1v[j] = e1;
    }
    // transpose P: C-layout -> A-layout through per-wave LDS tile
    _Float16* sl = pbuf[w];
#pragma unroll
    for (int j = 0; j < 8; ++j) {
      sl[(8 * hi + j) * 40 + n0] = (_Float16)p0v[j];
      sl[(8 * hi + j) * 40 + 16 + n0] = (_Float16)p1v[j];
    }
    asm volatile("s_wait_dscnt 0x0" ::: "memory");   // CDNA5 split DS counter
    v16h pa;
    {
      const _Float16* pr = sl + n0 * 40 + 16 * hi;
#pragma unroll
      for (int e = 0; e < 8; ++e) { pa[e] = pr[e]; pa[8 + e] = pr[8 + e]; }
    }
    // V as B-operand from LDS: B[k][d] = V[j0+k][d]
    v16h vb0, vb1;
    {
      const _Float16* pv = Vl + (16 * hi) * DH_ + n0;
#pragma unroll
      for (int e = 0; e < 16; ++e) { vb0[e] = pv[e * DH_]; vb1[e] = pv[e * DH_ + 16]; }
    }
    o0 = wmma_f16(pa, vb0, o0);
    o1 = wmma_f16(pa, vb1, o1);
    __syncthreads();                       // buffer free before re-DMA
  }
#pragma unroll
  for (int j = 0; j < 8; ++j) {
    float inv = l[j] > 0.f ? 1.f / l[j] : 0.f;
    int row = row0 + 8 * hi + j;
    float* op = outp + ((size_t)(b * N_ + row)) * D_ + head * DH_;
    op[n0] = o0[j] * inv;
    op[16 + n0] = o1[j] * inv;
  }
}

// ------------------------------ launch ------------------------------------
extern "C" void kernel_launch(void* const* d_in, const int* in_sizes, int n_in,
                              void* d_out, int out_size, void* d_ws, size_t ws_size,
                              hipStream_t stream) {
  const float* x    = (const float*)d_in[0];
  const float* xc   = (const float*)d_in[1];
  const int*   spd  = (const int*)d_in[2];
  const int*   mask = (const int*)d_in[3];
  const int*   eidx = (const int*)d_in[4];
  const float* eatt = (const float*)d_in[5];
  const float* in_w = (const float*)d_in[6];
  const float* in_b = (const float*)d_in[7];
  const float* ee_w1 = (const float*)d_in[8];
  const float* ee_b1 = (const float*)d_in[9];
  const float* ee_w2 = (const float*)d_in[10];
  const float* ee_b2 = (const float*)d_in[11];
  const float* demb  = (const float*)d_in[12];
  const float* q_w = (const float*)d_in[13]; const float* q_b = (const float*)d_in[14];
  const float* k_w = (const float*)d_in[15]; const float* k_b = (const float*)d_in[16];
  const float* v_w = (const float*)d_in[17]; const float* v_b = (const float*)d_in[18];
  const float* o_w = (const float*)d_in[19]; const float* o_b = (const float*)d_in[20];
  const float* f_w1 = (const float*)d_in[21]; const float* f_b1 = (const float*)d_in[22];
  const float* f_w2 = (const float*)d_in[23]; const float* f_b2 = (const float*)d_in[24];
  const float* ln1g = (const float*)d_in[25]; const float* ln1b = (const float*)d_in[26];
  const float* ln2g = (const float*)d_in[27]; const float* ln2b = (const float*)d_in[28];
  const float* ow   = (const float*)d_in[29]; const float* ob   = (const float*)d_in[30];

  char* ws = (char*)d_ws;
  size_t off = 0;
  auto alloc = [&](size_t bytes) -> void* {
    void* p = ws + off;
    off += (bytes + 255) & ~(size_t)255;
    return p;
  };
  float*    h   = (float*)alloc((size_t)BN_ * D_ * 4);
  float*    tA  = (float*)alloc((size_t)BN_ * D_ * 4);
  float*    tB  = (float*)alloc((size_t)BN_ * 2 * D_ * 4);
  float*    evb = (float*)alloc((size_t)B_ * E_ * H_ * 4);
  _Float16* qb  = (_Float16*)alloc((size_t)B_ * H_ * N_ * DH_ * 2);
  _Float16* kb  = (_Float16*)alloc((size_t)B_ * H_ * N_ * DH_ * 2);
  _Float16* vb  = (_Float16*)alloc((size_t)B_ * H_ * N_ * DH_ * 2);
  _Float16* bb  = (_Float16*)alloc((size_t)B_ * H_ * N_ * N_ * 2);
  if (off > ws_size) return;  // workspace too small; cannot run

  embed_kernel<<<BN_ * D_ / 256, 256, 0, stream>>>(x, xc, in_w, in_b, h);
  edge_vals<<<B_ * E_ / 8, 256, 0, stream>>>(eatt, ee_w1, ee_b1, ee_w2, ee_b2, evb);
  bias_base<<<(B_ * N_ * N_) / 256, 256, 0, stream>>>(spd, demb, bb);
  edge_scatter<<<(B_ * E_) / 256, 256, 0, stream>>>(eidx, spd, demb, evb, bb);

  const float scale = 0.17677669529663687f;  // dh^-0.5, dh=32
  for (int l = 0; l < L_; ++l) {
    gemm16<EPI_QKV><<<512, 256, 0, stream>>>(h, q_w + (size_t)l * D_ * D_,
                                             q_b + l * D_, qb, D_, D_);
    gemm16<EPI_QKV><<<512, 256, 0, stream>>>(h, k_w + (size_t)l * D_ * D_,
                                             k_b + l * D_, kb, D_, D_);
    gemm16<EPI_QKV><<<512, 256, 0, stream>>>(h, v_w + (size_t)l * D_ * D_,
                                             v_b + l * D_, vb, D_, D_);
    flash_attn<<<B_ * H_ * 8, 128, 0, stream>>>(qb, kb, vb, bb, mask, tA, scale);
    gemm16<EPI_STORE><<<512, 256, 0, stream>>>(tA, o_w + (size_t)l * D_ * D_,
                                               o_b + l * D_, tB, D_, D_);
    ln_res<<<BN_ / 8, 256, 0, stream>>>(tB, h, ln1g + l * D_, ln1b + l * D_);
    gemm16<EPI_RELU><<<1024, 256, 0, stream>>>(h, f_w1 + (size_t)l * D_ * 2 * D_,
                                               f_b1 + l * 2 * D_, tB, D_, 2 * D_);
    gemm16<EPI_STORE><<<512, 256, 0, stream>>>(tB, f_w2 + (size_t)l * 2 * D_ * D_,
                                               f_b2 + l * D_, tA, 2 * D_, D_);
    ln_res<<<BN_ / 8, 256, 0, stream>>>(tA, h, ln2g + l * D_, ln2b + l * D_);
  }
  mask_copy<<<BN_ * D_ / 256, 256, 0, stream>>>(h, mask, tA);
  gemm16<EPI_STORE><<<128, 256, 0, stream>>>(tA, ow, ob, (float*)d_out, D_, 64);
}
